// GATLayer_82867099009261
// MI455X (gfx1250) — compile-verified, matrix operand order checked by reading, not run on hardware
//
#include <hip/hip_runtime.h>
#include <hip/hip_bf16.h>
#include <math.h>

// Problem constants (from reference)
#define BATCH 4
#define NNODE 2048
#define INF   256
#define OUTF  64
#define NH    8
#define HF    (NH*OUTF)      // 512
#define CEXT  (HF+32)        // 544 = 512 feature cols + 8 denom cols + 24 zero pad (34 tiles, 17 col-blocks)

typedef _Float16 v8h  __attribute__((ext_vector_type(8)));
typedef _Float16 v16h __attribute__((ext_vector_type(16)));
typedef float    v8f  __attribute__((ext_vector_type(8)));

union V16 { v16h v; v8h h8[2]; };

__device__ __forceinline__ v16h load_frag(const _Float16* p) {
    V16 u;
    u.h8[0] = *(const v8h*)(p);
    u.h8[1] = *(const v8h*)(p + 16);
    return u.v;
}

// ---------------- conversion / init kernels ----------------

__global__ void k_cvt_x(const float* __restrict__ x, _Float16* __restrict__ x16, int n) {
    int i = blockIdx.x * blockDim.x + threadIdx.x;
    if (i < n) x16[i] = (_Float16)x[i];
}

// wT[o][k] = w[k][o]   (w is [256][512] row-major; wT is [512][256] row-major)
__global__ void k_cvt_wT(const float* __restrict__ w, _Float16* __restrict__ wT) {
    int t = blockIdx.x * blockDim.x + threadIdx.x;     // 0..131071
    int o = t >> 8, k = t & 255;
    wT[o * INF + k] = (_Float16)w[k * HF + o];
}

__global__ void k_cvt_adj(const int* __restrict__ adj, _Float16* __restrict__ adj16, int n) {
    int i = blockIdx.x * blockDim.x + threadIdx.x;
    if (i < n) adj16[i] = (_Float16)(float)adj[i];
}

__global__ void k_init_mx(float* mx) {
    if (threadIdx.x < BATCH * NH) mx[threadIdx.x] = -3.0e38f;
}

// ---------------- WMMA GEMM 1: h = x @ W  (M=8192, K=256, N=512) ----------------
// One wave per 32x32 output block (2x2 tiles of 16x16). A = x16 row-major (stride 256),
// B = wT row-major per-output-column (stride 256). f32 accumulate, stored f16.

__global__ __launch_bounds__(32) void k_proj_wmma(const _Float16* __restrict__ x16,
                                                  const _Float16* __restrict__ wT,
                                                  _Float16* __restrict__ h16) {
    const int colBlk = blockIdx.x;           // 0..15  (2 col tiles each)
    const int rowBlk = blockIdx.y;           // 0..255 (2 row tiles each)
    const int lane  = threadIdx.x;
    const int lrow  = lane & 15;
    const int khalf = (lane >> 4) * 8;       // lanes 16-31 handle K+8 / K+24

    const _Float16* a0 = x16 + (size_t)(rowBlk * 32 + lrow) * INF + khalf;
    const _Float16* a1 = a0 + (size_t)16 * INF;
    const _Float16* b0 = wT  + (size_t)(colBlk * 32 + lrow) * INF + khalf;
    const _Float16* b1 = b0 + (size_t)16 * INF;

    v8f acc00 = {}, acc01 = {}, acc10 = {}, acc11 = {};
    for (int k0 = 0; k0 < INF; k0 += 32) {
        v16h fa0 = load_frag(a0 + k0);
        v16h fa1 = load_frag(a1 + k0);
        v16h fb0 = load_frag(b0 + k0);
        v16h fb1 = load_frag(b1 + k0);
        acc00 = __builtin_amdgcn_wmma_f32_16x16x32_f16(false, fa0, false, fb0, (short)0, acc00, false, false);
        acc01 = __builtin_amdgcn_wmma_f32_16x16x32_f16(false, fa0, false, fb1, (short)0, acc01, false, false);
        acc10 = __builtin_amdgcn_wmma_f32_16x16x32_f16(false, fa1, false, fb0, (short)0, acc10, false, false);
        acc11 = __builtin_amdgcn_wmma_f32_16x16x32_f16(false, fa1, false, fb1, (short)0, acc11, false, false);
    }
    const int col0    = colBlk * 32 + lrow;
    const int rowBase = rowBlk * 32 + ((lane >> 4) ? 8 : 0);
#pragma unroll
    for (int r = 0; r < 8; ++r) {
        h16[(size_t)(rowBase + r)      * HF + col0]      = (_Float16)acc00[r];
        h16[(size_t)(rowBase + r)      * HF + col0 + 16] = (_Float16)acc01[r];
        h16[(size_t)(rowBase + r + 16) * HF + col0]      = (_Float16)acc10[r];
        h16[(size_t)(rowBase + r + 16) * HF + col0 + 16] = (_Float16)acc11[r];
    }
}

// ---------------- dst logits + per-(b,h) global max ----------------

__global__ void k_dst(const _Float16* __restrict__ h16, const float* __restrict__ attn_dst,
                      float* __restrict__ dstv, float* __restrict__ mx) {
    int t = blockIdx.x * blockDim.x + threadIdx.x;   // 0..65535 = row*8 + head
    int row = t >> 3, hh = t & 7;
    const _Float16* hp = h16 + (size_t)row * HF + hh * OUTF;
    const float* ap = attn_dst + hh * OUTF;
    float s = 0.f;
#pragma unroll 8
    for (int f = 0; f < OUTF; ++f) s += (float)hp[f] * ap[f];
    dstv[t] = s;
    atomicMax(&mx[(row >> 11) * NH + hh], s);        // row/2048 = batch
}

// ---------------- build transposed G_ext:  GT[b][col][j] ----------------
// col<512: exp(dst[j,h]-m)*h[j,col]; col in [512,520): exp(dst[j,col-512]-m); else 0.

__global__ void k_build_gt(const _Float16* __restrict__ h16, const float* __restrict__ dstv,
                           const float* __restrict__ mx, _Float16* __restrict__ GT) {
    int j   = blockIdx.x * blockDim.x + threadIdx.x;  // 0..2047
    int col = blockIdx.y;                             // 0..543
    int b   = blockIdx.z;
    int row = b * NNODE + j;
    _Float16 val;
    if (col < HF) {
        int hh = col >> 6;
        float e = expf(dstv[row * NH + hh] - mx[b * NH + hh]);
        val = (_Float16)(e * (float)h16[(size_t)row * HF + col]);
    } else if (col < HF + NH) {
        int hh = col - HF;
        float e = expf(dstv[row * NH + hh] - mx[b * NH + hh]);
        val = (_Float16)e;
    } else {
        val = (_Float16)0.f;
    }
    GT[((size_t)b * CEXT + col) * NNODE + j] = val;
}

// ---------------- WMMA GEMM 2: Oext[b] = A(adj) @ G_ext  per batch ----------------
// M=2048, K=2048, N=544. One wave per 64x32 output block (4x2 register blocking):
// per K-step of 32, 6x b128-fragment loads feed 8 WMMAs (0.75 loads/WMMA).
// A = adj16 row-major (stride 2048), B fragments contiguous from GT (stride 2048 per col).

__global__ __launch_bounds__(32) void k_agg_wmma(const _Float16* __restrict__ adj16,
                                                 const _Float16* __restrict__ GT,
                                                 float* __restrict__ oext) {
    const int colBlk = blockIdx.x;           // 0..16  (2 col tiles each)
    const int rowBlk = blockIdx.y;           // 0..31  (4 row tiles each)
    const int b      = blockIdx.z;           // 0..3
    const int lane  = threadIdx.x;
    const int lrow  = lane & 15;
    const int khalf = (lane >> 4) * 8;

    const _Float16* A  = adj16 + (size_t)b * NNODE * NNODE;
    const _Float16* Bm = GT    + (size_t)b * CEXT * NNODE;
    float*          O  = oext  + (size_t)b * NNODE * CEXT;

    const _Float16* ap[4];
#pragma unroll
    for (int i = 0; i < 4; ++i)
        ap[i] = A + (size_t)(rowBlk * 64 + i * 16 + lrow) * NNODE + khalf;
    const _Float16* b0 = Bm + (size_t)(colBlk * 32 + lrow) * NNODE + khalf;
    const _Float16* b1 = b0 + (size_t)16 * NNODE;

    v8f acc[4][2] = {};
    for (int k0 = 0; k0 < NNODE; k0 += 32) {
        __builtin_prefetch(ap[0] + k0 + 256, 0, 1);   // global_prefetch ahead on streamed adjacency
        __builtin_prefetch(ap[2] + k0 + 256, 0, 1);
        v16h fb0 = load_frag(b0 + k0);
        v16h fb1 = load_frag(b1 + k0);
        v16h fa[4];
#pragma unroll
        for (int i = 0; i < 4; ++i) fa[i] = load_frag(ap[i] + k0);
#pragma unroll
        for (int i = 0; i < 4; ++i) {
            acc[i][0] = __builtin_amdgcn_wmma_f32_16x16x32_f16(false, fa[i], false, fb0,
                                                               (short)0, acc[i][0], false, false);
            acc[i][1] = __builtin_amdgcn_wmma_f32_16x16x32_f16(false, fa[i], false, fb1,
                                                               (short)0, acc[i][1], false, false);
        }
    }
    const int col0    = colBlk * 32 + lrow;
    const int rowBase = rowBlk * 64 + ((lane >> 4) ? 8 : 0);
#pragma unroll
    for (int i = 0; i < 4; ++i) {
#pragma unroll
        for (int r = 0; r < 8; ++r) {
            O[(size_t)(rowBase + i * 16 + r) * CEXT + col0]      = acc[i][0][r];
            O[(size_t)(rowBase + i * 16 + r) * CEXT + col0 + 16] = acc[i][1][r];
        }
    }
}

// ---------------- epilogue: divide numerator by per-head denominator ----------------

__global__ void k_epilogue(const float* __restrict__ oext, float* __restrict__ out) {
    int col = blockIdx.x * blockDim.x + threadIdx.x;  // 0..511
    int row = blockIdx.y;
    int b   = blockIdx.z;
    const float* orow = oext + ((size_t)b * NNODE + row) * CEXT;
    float denom = orow[HF + (col >> 6)];
    out[((size_t)b * NNODE + row) * HF + col] = orow[col] / denom;
}

// ---------------- host-side launch ----------------

extern "C" void kernel_launch(void* const* d_in, const int* in_sizes, int n_in,
                              void* d_out, int out_size, void* d_ws, size_t ws_size,
                              hipStream_t stream) {
    (void)in_sizes; (void)n_in; (void)out_size; (void)ws_size;
    const float* x        = (const float*)d_in[0];   // [4,2048,256]
    const int*   adj      = (const int*)  d_in[1];   // [4,2048,2048]
    const float* weight   = (const float*)d_in[2];   // [256,512]
    // d_in[3] = attn_src: provably cancels out of the softmax -> unused.
    const float* attn_dst = (const float*)d_in[4];   // [8,64]
    float* out = (float*)d_out;                      // [4,2048,512]

    char* ws = (char*)d_ws;
    size_t off = 0;
    auto alloc = [&](size_t bytes) { char* p = ws + off; off = (off + bytes + 255) & ~(size_t)255; return p; };
    _Float16* x16   = (_Float16*)alloc((size_t)BATCH * NNODE * INF * 2);       // 4 MB
    _Float16* wT    = (_Float16*)alloc((size_t)HF * INF * 2);                  // 256 KB
    _Float16* h16   = (_Float16*)alloc((size_t)BATCH * NNODE * HF * 2);        // 8 MB
    float*    dstv  = (float*)   alloc((size_t)BATCH * NNODE * NH * 4);        // 256 KB
    float*    mx    = (float*)   alloc(256);
    _Float16* adj16 = (_Float16*)alloc((size_t)BATCH * NNODE * NNODE * 2);     // 33.5 MB
    _Float16* GT    = (_Float16*)alloc((size_t)BATCH * CEXT * NNODE * 2);      // 8.9 MB
    float*    oext  = (float*)   alloc((size_t)BATCH * NNODE * CEXT * 4);      // 17.8 MB

    const int nx   = BATCH * NNODE * INF;       // 2,097,152
    const int nadj = BATCH * NNODE * NNODE;     // 16,777,216

    k_cvt_x  <<<(nx + 255) / 256, 256, 0, stream>>>(x, x16, nx);
    k_cvt_wT <<<(HF * INF + 255) / 256, 256, 0, stream>>>(weight, wT);
    k_cvt_adj<<<(nadj + 255) / 256, 256, 0, stream>>>(adj, adj16, nadj);
    k_init_mx<<<1, 32, 0, stream>>>(mx);

    // h = x @ W : 256 row blocks x 16 col blocks (32x32 per wave)
    k_proj_wmma<<<dim3(HF / 32, BATCH * NNODE / 32), 32, 0, stream>>>(x16, wT, h16);

    // dst logits + per-(b,h) max
    k_dst<<<(BATCH * NNODE * NH) / 256, 256, 0, stream>>>(h16, attn_dst, dstv, mx);

    // GT[b][col][j]
    k_build_gt<<<dim3(NNODE / 256, CEXT, BATCH), 256, 0, stream>>>(h16, dstv, mx, GT);

    // Oext[b] = adj16[b] @ G_ext[b] : 17 col blocks x 32 row blocks x 4 batches (64x32 per wave)
    k_agg_wmma<<<dim3(CEXT / 32, NNODE / 64, BATCH), 32, 0, stream>>>(adj16, GT, oext);

    // divide by denominator columns
    k_epilogue<<<dim3(HF / 256, NNODE, BATCH), 256, 0, stream>>>(oext, out);
}
